// PIGN_32512902431147
// MI455X (gfx1250) — matrix operand order repaired and checked by reading, not compiled, not run on hardware
//
#include <hip/hip_runtime.h>

typedef __attribute__((ext_vector_type(16))) _Float16 v16h;
typedef __attribute__((ext_vector_type(8)))  float    v8f;

#define NN 50000
#define EE 1600000
#define GG 8
#define FXC 13
#define FPC 3

// ---------------- WMMA fragment helpers (CDNA5 16x16x32 f16, wave32) -----------

__device__ __forceinline__ float nf_val(const float* __restrict__ x,
                                        const float* __restrict__ pos,
                                        int i, int c) {
    return (c < FXC) ? x[i * FXC + c] : pos[i * FPC + (c - FXC)];
}

// A tile stored row-major [16][64] f16 in LDS.
// 16-bit A-matrix 16x32 layout: lanes 0-15 hold K {0..7,16..23}, lanes 16-31 hold +8.
__device__ __forceinline__ v16h load_a_frag(const _Float16* __restrict__ tile,
                                            int kbase, int lane) {
    int m   = lane & 15;
    int klo = (lane >> 4) << 3;            // 0 or 8
    const _Float16* row = tile + m * 64 + kbase + klo;
    v16h a;
#pragma unroll
    for (int h = 0; h < 8; ++h)  a[h] = row[h];        // K = klo + 0..7
#pragma unroll
    for (int h = 8; h < 16; ++h) a[h] = row[h + 8];    // K = klo + 16..23
    return a;
}

// Weights staged TRANSPOSED in LDS: wt[n*64 + k]  (so B-frag reads are contiguous in K)
// B 32x16 layout: N = lane&15, lanes 0-15 hold K {0..7,16..23}, lanes 16-31 hold +8.
__device__ __forceinline__ v16h load_b_frag(const _Float16* __restrict__ wt,
                                            int kbase, int nbase, int lane) {
    int n  = nbase + (lane & 15);
    int ko = (lane >> 4) << 3;             // 0 or 8
    const _Float16* colp = wt + n * 64 + kbase + ko;
    v16h b;
#pragma unroll
    for (int h = 0; h < 16; ++h) b[h] = colp[h < 8 ? h : h + 8];
    return b;
}

__device__ __forceinline__ v8f wmma32(v16h a, v16h b, v8f c) {
    return __builtin_amdgcn_wmma_f32_16x16x32_f16(false, a, false, b,
                                                  (short)0, c, false, false);
}

// ------------------------------- Edge kernel ----------------------------------
// 8 waves/block, 16 edges/wave.  e_in = [nf[src] | nf[dst] | edge_attr | g[eb]] (K=64)
__global__ __launch_bounds__(256) void edge_kernel(
    const float* __restrict__ x, const float* __restrict__ pos,
    const float* __restrict__ ea, const float* __restrict__ gf,
    const int* __restrict__ ei, const int* __restrict__ batch,
    const float* __restrict__ W1, const float* __restrict__ b1,
    const float* __restrict__ W2, const float* __restrict__ b2,
    float* __restrict__ out_ef,
    float* __restrict__ nodeSum, float* __restrict__ nodeCnt,
    float* __restrict__ gESum, float* __restrict__ gECnt)
{
    __shared__ _Float16 sW1t[64 * 64];     // transposed [n][k]
    __shared__ _Float16 sW2t[16 * 64];     // transposed [n][k]
    __shared__ float    sb1[64];
    __shared__ float    sb2[16];
    __shared__ _Float16 sA[8][16 * 64];    // per-wave activation tile
    __shared__ int      sDst[8][16];
    __shared__ int      sGb[8][16];
    __shared__ float    sGacc[GG * 16];    // block-level per-graph partial sums
    __shared__ float    sGcnt[GG];

    const int tid = threadIdx.x;
    for (int i = tid; i < 64 * 64; i += 256) {
        int k = i >> 6, n = i & 63;
        sW1t[n * 64 + k] = (_Float16)W1[i];            // W1 is [64][64] row-major
    }
    for (int i = tid; i < 64 * 16; i += 256) {
        int k = i >> 4, n = i & 15;
        sW2t[n * 64 + k] = (_Float16)W2[i];            // W2 is [64][16]
    }
    if (tid < 64) sb1[tid] = b1[tid];
    if (tid < 16) sb2[tid] = b2[tid];
    if (tid < GG * 16) sGacc[tid] = 0.f;
    if (tid < GG) sGcnt[tid] = 0.f;
    __syncthreads();

    const int w = tid >> 5, lane = tid & 31;
    const int e0 = (blockIdx.x * 8 + w) * 16;
    _Float16* A = &sA[w][0];

    { // build tile: 2 lanes per row
        int r = lane & 15, part = lane >> 4;
        int e = e0 + r;
        if (part == 0) {
            int s = ei[e], d = ei[EE + e];
            sDst[w][r] = d;
            sGb[w][r]  = batch[s];
#pragma unroll
            for (int c = 0; c < 16; ++c) {
                A[r * 64 + c]      = (_Float16)nf_val(x, pos, s, c);
                A[r * 64 + 16 + c] = (_Float16)nf_val(x, pos, d, c);
            }
        } else {
            int s = ei[e];
            int gb = batch[s];
#pragma unroll
            for (int c = 0; c < 16; ++c) {
                A[r * 64 + 32 + c] = (_Float16)ea[e * 16 + c];
                A[r * 64 + 48 + c] = (_Float16)gf[gb * 16 + c];
            }
        }
    }
    __syncthreads();

    // layer 1: [16x64] @ [64x64]
    v8f c0 = {}, c1 = {}, c2 = {}, c3 = {};
#pragma unroll
    for (int s = 0; s < 2; ++s) {
        v16h a = load_a_frag(A, 32 * s, lane);
        c0 = wmma32(a, load_b_frag(sW1t, 32 * s,  0, lane), c0);
        c1 = wmma32(a, load_b_frag(sW1t, 32 * s, 16, lane), c1);
        c2 = wmma32(a, load_b_frag(sW1t, 32 * s, 32, lane), c2);
        c3 = wmma32(a, load_b_frag(sW1t, 32 * s, 48, lane), c3);
    }
    __syncthreads();

    const int col  = lane & 15;
    const int roff = (lane >> 4) << 3;
    v8f cc[4] = {c0, c1, c2, c3};
#pragma unroll
    for (int t = 0; t < 4; ++t) {
        float bb = sb1[t * 16 + col];
#pragma unroll
        for (int v = 0; v < 8; ++v) {
            float hv = cc[t][v] + bb;
            hv = hv > 0.f ? hv : 0.f;                  // ReLU
            A[(roff + v) * 64 + t * 16 + col] = (_Float16)hv;
        }
    }
    __syncthreads();

    // layer 2: [16x64] @ [64x16]
    v8f d = {};
#pragma unroll
    for (int s = 0; s < 2; ++s) {
        v16h a = load_a_frag(A, 32 * s, lane);
        d = wmma32(a, load_b_frag(sW2t, 32 * s, 0, lane), d);
    }

    float bb2 = sb2[col];
#pragma unroll
    for (int v = 0; v < 8; ++v) {
        int r = roff + v;
        int e = e0 + r;
        float val = d[v] + bb2;
        out_ef[e * 16 + col] = val;
        atomicAdd(&nodeSum[sDst[w][r] * 16 + col], val);       // low-contention L2 atomic
        atomicAdd(&sGacc[sGb[w][r] * 16 + col], val);          // hot: stage in LDS
    }
    if (lane < 16) {
        atomicAdd(&nodeCnt[sDst[w][lane]], 1.0f);
        atomicAdd(&sGcnt[sGb[w][lane]], 1.0f);
    }
    __syncthreads();

    // one global atomic per element per block for the hot per-graph accumulators
    if (tid < GG * 16) atomicAdd(&gESum[tid], sGacc[tid]);
    if (tid < GG)      atomicAdd(&gECnt[tid], sGcnt[tid]);
}

// ------------------------------- Node kernel ----------------------------------
// n_in = [agg_ef | nf | g[batch] | 0-pad] (K=48 -> 64), + residual nf
__global__ __launch_bounds__(256) void node_kernel(
    const float* __restrict__ x, const float* __restrict__ pos,
    const float* __restrict__ gf, const int* __restrict__ batch,
    const float* __restrict__ W1, const float* __restrict__ b1,
    const float* __restrict__ W2, const float* __restrict__ b2,
    const float* __restrict__ nodeSum, const float* __restrict__ nodeCnt,
    float* __restrict__ out_nf,
    float* __restrict__ gNSum, float* __restrict__ gNCnt, int ntiles)
{
    __shared__ _Float16 sW1t[64 * 64];
    __shared__ _Float16 sW2t[16 * 64];
    __shared__ float    sb1[64];
    __shared__ float    sb2[16];
    __shared__ _Float16 sA[8][16 * 64];
    __shared__ int      sBat[8][16];
    __shared__ float    sGacc[GG * 16];
    __shared__ float    sGcnt[GG];

    const int tid = threadIdx.x;
    for (int i = tid; i < 64 * 64; i += 256) {
        int k = i >> 6, n = i & 63;
        sW1t[n * 64 + k] = (k < 48) ? (_Float16)W1[k * 64 + n] : (_Float16)0.f;
    }
    for (int i = tid; i < 64 * 16; i += 256) {
        int k = i >> 4, n = i & 15;
        sW2t[n * 64 + k] = (_Float16)W2[i];
    }
    if (tid < 64) sb1[tid] = b1[tid];
    if (tid < 16) sb2[tid] = b2[tid];
    if (tid < GG * 16) sGacc[tid] = 0.f;
    if (tid < GG) sGcnt[tid] = 0.f;
    __syncthreads();

    const int w = tid >> 5, lane = tid & 31;
    const int tile = blockIdx.x * 8 + w;
    const bool valid = tile < ntiles;
    const int n0 = (valid ? tile : 0) * 16;
    _Float16* A = &sA[w][0];

    {
        int r = lane & 15, part = lane >> 4;
        int node = n0 + r;                 // always in range (ntiles*16 == NN)
        if (part == 0) {
            float cnt = nodeCnt[node];
            float inv = 1.0f / (cnt > 1.0f ? cnt : 1.0f);
            sBat[w][r] = batch[node];
#pragma unroll
            for (int c = 0; c < 16; ++c) {
                A[r * 64 + c]      = (_Float16)(nodeSum[node * 16 + c] * inv);
                A[r * 64 + 16 + c] = (_Float16)nf_val(x, pos, node, c);
            }
        } else {
            int gb = batch[node];
#pragma unroll
            for (int c = 0; c < 16; ++c) {
                A[r * 64 + 32 + c] = (_Float16)gf[gb * 16 + c];
                A[r * 64 + 48 + c] = (_Float16)0.f;    // K pad 48..63
            }
        }
    }
    __syncthreads();

    v8f c0 = {}, c1 = {}, c2 = {}, c3 = {};
#pragma unroll
    for (int s = 0; s < 2; ++s) {
        v16h a = load_a_frag(A, 32 * s, lane);
        c0 = wmma32(a, load_b_frag(sW1t, 32 * s,  0, lane), c0);
        c1 = wmma32(a, load_b_frag(sW1t, 32 * s, 16, lane), c1);
        c2 = wmma32(a, load_b_frag(sW1t, 32 * s, 32, lane), c2);
        c3 = wmma32(a, load_b_frag(sW1t, 32 * s, 48, lane), c3);
    }
    __syncthreads();

    const int col  = lane & 15;
    const int roff = (lane >> 4) << 3;
    v8f cc[4] = {c0, c1, c2, c3};
#pragma unroll
    for (int t = 0; t < 4; ++t) {
        float bb = sb1[t * 16 + col];
#pragma unroll
        for (int v = 0; v < 8; ++v) {
            float hv = cc[t][v] + bb;
            hv = hv > 0.f ? hv : 0.f;
            A[(roff + v) * 64 + t * 16 + col] = (_Float16)hv;
        }
    }
    __syncthreads();

    v8f d = {};
#pragma unroll
    for (int s = 0; s < 2; ++s) {
        v16h a = load_a_frag(A, 32 * s, lane);
        d = wmma32(a, load_b_frag(sW2t, 32 * s, 0, lane), d);
    }

    if (valid) {                            // wave-uniform guard
        float bb2 = sb2[col];
#pragma unroll
        for (int v = 0; v < 8; ++v) {
            int r = roff + v;
            int node = n0 + r;
            float val = d[v] + bb2 + nf_val(x, pos, node, col);   // residual
            out_nf[node * 16 + col] = val;
            atomicAdd(&sGacc[sBat[w][r] * 16 + col], val);        // stage in LDS
        }
        if (lane < 16) atomicAdd(&sGcnt[sBat[w][lane]], 1.0f);
    }
    __syncthreads();

    if (tid < GG * 16) atomicAdd(&gNSum[tid], sGacc[tid]);
    if (tid < GG)      atomicAdd(&gNCnt[tid], sGcnt[tid]);
}

// ------------------------------ Global kernel ---------------------------------
// g_in = [agg_nf | agg_ef | u | 0-pad] (8 rows padded to 16), + residual u
__global__ __launch_bounds__(32) void global_kernel(
    const float* __restrict__ gf,
    const float* __restrict__ W1, const float* __restrict__ b1,
    const float* __restrict__ W2, const float* __restrict__ b2,
    const float* __restrict__ gNSum, const float* __restrict__ gNCnt,
    const float* __restrict__ gESum, const float* __restrict__ gECnt,
    float* __restrict__ out_u)
{
    __shared__ _Float16 sW1t[64 * 64];
    __shared__ _Float16 sW2t[16 * 64];
    __shared__ float    sb1[64];
    __shared__ float    sb2[16];
    __shared__ _Float16 A[16 * 64];

    const int lane = threadIdx.x;
    for (int i = lane; i < 64 * 64; i += 32) {
        int k = i >> 6, n = i & 63;
        sW1t[n * 64 + k] = (k < 48) ? (_Float16)W1[k * 64 + n] : (_Float16)0.f;
    }
    for (int i = lane; i < 64 * 16; i += 32) {
        int k = i >> 4, n = i & 15;
        sW2t[n * 64 + k] = (_Float16)W2[i];
    }
    for (int i = lane; i < 64; i += 32) sb1[i] = b1[i];
    if (lane < 16) sb2[lane] = b2[lane];
    __syncthreads();

    {
        int r = lane & 15, part = lane >> 4;
        if (part == 0) {
#pragma unroll
            for (int c = 0; c < 16; ++c) {
                float a0 = 0.f, a1 = 0.f;
                if (r < GG) {
                    float cn = gNCnt[r]; cn = cn > 1.f ? cn : 1.f;
                    float ce = gECnt[r]; ce = ce > 1.f ? ce : 1.f;
                    a0 = gNSum[r * 16 + c] / cn;
                    a1 = gESum[r * 16 + c] / ce;
                }
                A[r * 64 + c]      = (_Float16)a0;
                A[r * 64 + 16 + c] = (_Float16)a1;
            }
        } else {
#pragma unroll
            for (int c = 0; c < 16; ++c) {
                float u = (r < GG) ? gf[r * 16 + c] : 0.f;
                A[r * 64 + 32 + c] = (_Float16)u;
                A[r * 64 + 48 + c] = (_Float16)0.f;
            }
        }
    }
    __syncthreads();

    v8f c0 = {}, c1 = {}, c2 = {}, c3 = {};
#pragma unroll
    for (int s = 0; s < 2; ++s) {
        v16h a = load_a_frag(A, 32 * s, lane);
        c0 = wmma32(a, load_b_frag(sW1t, 32 * s,  0, lane), c0);
        c1 = wmma32(a, load_b_frag(sW1t, 32 * s, 16, lane), c1);
        c2 = wmma32(a, load_b_frag(sW1t, 32 * s, 32, lane), c2);
        c3 = wmma32(a, load_b_frag(sW1t, 32 * s, 48, lane), c3);
    }
    __syncthreads();

    const int col  = lane & 15;
    const int roff = (lane >> 4) << 3;
    v8f cc[4] = {c0, c1, c2, c3};
#pragma unroll
    for (int t = 0; t < 4; ++t) {
        float bb = sb1[t * 16 + col];
#pragma unroll
        for (int v = 0; v < 8; ++v) {
            float hv = cc[t][v] + bb;
            hv = hv > 0.f ? hv : 0.f;
            A[(roff + v) * 64 + t * 16 + col] = (_Float16)hv;
        }
    }
    __syncthreads();

    v8f d = {};
#pragma unroll
    for (int s = 0; s < 2; ++s) {
        v16h a = load_a_frag(A, 32 * s, lane);
        d = wmma32(a, load_b_frag(sW2t, 32 * s, 0, lane), d);
    }

    if (lane < 16) {                        // rows 0..7 live in lanes 0..15
#pragma unroll
        for (int v = 0; v < 8; ++v)
            out_u[v * 16 + col] = d[v] + sb2[col] + gf[v * 16 + col];   // residual
    }
}

// ------------------------------- Launcher -------------------------------------
extern "C" void kernel_launch(void* const* d_in, const int* in_sizes, int n_in,
                              void* d_out, int out_size, void* d_ws, size_t ws_size,
                              hipStream_t stream) {
    const float* x     = (const float*)d_in[0];
    const float* pos   = (const float*)d_in[1];
    const float* ea    = (const float*)d_in[2];
    const float* gf    = (const float*)d_in[3];
    const int*   ei    = (const int*)  d_in[4];
    const int*   batch = (const int*)  d_in[5];
    const float* eW1 = (const float*)d_in[6],  *eb1 = (const float*)d_in[7];
    const float* eW2 = (const float*)d_in[8],  *eb2 = (const float*)d_in[9];
    const float* nW1 = (const float*)d_in[10], *nb1 = (const float*)d_in[11];
    const float* nW2 = (const float*)d_in[12], *nb2 = (const float*)d_in[13];
    const float* gW1 = (const float*)d_in[14], *gb1 = (const float*)d_in[15];
    const float* gW2 = (const float*)d_in[16], *gb2 = (const float*)d_in[17];

    float* out   = (float*)d_out;
    float* o_nf  = out;                                   // [N,16]
    float* o_ef  = out + (size_t)NN * 16;                 // [E,16]
    float* o_u   = o_ef + (size_t)EE * 16;                // [G,16]

    float* ws      = (float*)d_ws;
    float* nodeSum = ws;                                  // N*16
    float* nodeCnt = nodeSum + (size_t)NN * 16;           // N
    float* gESum   = nodeCnt + NN;                        // G*16
    float* gECnt   = gESum + GG * 16;                     // G
    float* gNSum   = gECnt + GG;                          // G*16
    float* gNCnt   = gNSum + GG * 16;                     // G
    size_t ws_floats = (size_t)NN * 16 + NN + 2 * (GG * 16 + GG);
    hipMemsetAsync(d_ws, 0, ws_floats * sizeof(float), stream);

    // Edge MLP + scatters: E/128 = 12500 full blocks
    edge_kernel<<<EE / 128, 256, 0, stream>>>(
        x, pos, ea, gf, ei, batch, eW1, eb1, eW2, eb2,
        o_ef, nodeSum, nodeCnt, gESum, gECnt);

    // Node MLP: N/16 = 3125 tiles, 8 tiles/block
    int ntiles = NN / 16;
    node_kernel<<<(ntiles + 7) / 8, 256, 0, stream>>>(
        x, pos, gf, batch, nW1, nb1, nW2, nb2,
        nodeSum, nodeCnt, o_nf, gNSum, gNCnt, ntiles);

    // Global MLP: one wave
    global_kernel<<<1, 32, 0, stream>>>(
        gf, gW1, gb1, gW2, gb2, gNSum, gNCnt, gESum, gECnt, o_u);
}
